// MixtureOfExperts_41180146434508
// MI455X (gfx1250) — matrix operand order repaired and checked by reading, not compiled, not run on hardware
//
#include <hip/hip_runtime.h>
#include <hip/hip_bf16.h>

typedef __attribute__((ext_vector_type(16))) _Float16 v16h;
typedef __attribute__((ext_vector_type(8)))  _Float16 v8h;
typedef __attribute__((ext_vector_type(8)))  float    v8f;

#define D_DIM 1024
#define H_DIM 4096
#define NEXP  8
#define MT    32      // tokens per FFN tile (2 row-groups of WMMA M=16)
#define HC    256     // H chunk resident in LDS between phase1/phase2
#define NW    8       // waves per 256-thread block (wave32)

// dynamic LDS layout (bytes): xs[MT][D_DIM] f16 (64KB) | hs[MT][HC] f16 (16KB) | s_tok[MT] | s_w[MT]
#define SMEM_BYTES ((MT * D_DIM + MT * HC) * 2 + MT * 4 + MT * 4)

__device__ __forceinline__ float gelu_f(float v) {
    const float c0 = 0.7978845608028654f;   // sqrt(2/pi)
    const float c1 = 0.044715f;
    float u = c0 * (v + c1 * v * v * v);
    // branch-free tanh: tanh(u) = 1 - 2/(e^{2u}+1); exact limits at +-inf
    float ex = __expf(2.0f * u);            // v_exp_f32
    float th = 1.0f - 2.0f * __builtin_amdgcn_rcpf(ex + 1.0f);
    return 0.5f * v * (1.0f + th);          // jax.nn.gelu (approximate=True)
}

// A-fragment (16x32 f16) from row-major LDS [rows][stride]:
// lane l holds row m; K group: [k0+kh*16, +16) -> a[0..15]
__device__ __forceinline__ v16h load_a_frag(const _Float16* base, int stride,
                                            int m, int kh, int k0) {
    const _Float16* p = base + m * stride + k0 + kh * 16;
    v8h lo = *(const v8h*)p;          // 16B LDS load
    v8h hi = *(const v8h*)(p + 8);    // 16B LDS load
    v16h a;
#pragma unroll
    for (int i = 0; i < 8; ++i) { a[i] = lo[i]; a[i + 8] = hi[i]; }
    return a;
}

// ---------------- gating: logits -> softmax -> top2 -> expert lists ----------------
__global__ __launch_bounds__(256) void moe_gate_kernel(
    const float* __restrict__ x, const float* __restrict__ Wg,
    int* __restrict__ cnt, int* __restrict__ toks, float* __restrict__ wts, int T)
{
    const int wave = threadIdx.x >> 5;
    const int lane = threadIdx.x & 31;
    const int t = blockIdx.x * NW + wave;
    if (t >= T) return;

    float acc[NEXP];
#pragma unroll
    for (int e = 0; e < NEXP; ++e) acc[e] = 0.0f;
    for (int d = lane; d < D_DIM; d += 32) {
        const float xv = x[(size_t)t * D_DIM + d];
        const float* wr = Wg + d * NEXP;          // Wg[d][0..7] contiguous
#pragma unroll
        for (int e = 0; e < NEXP; ++e) acc[e] = fmaf(xv, wr[e], acc[e]);
    }
#pragma unroll
    for (int off = 16; off > 0; off >>= 1)
#pragma unroll
        for (int e = 0; e < NEXP; ++e) acc[e] += __shfl_xor(acc[e], off, 32);

    if (lane == 0) {
        float mx = acc[0];
        for (int e = 1; e < NEXP; ++e) mx = fmaxf(mx, acc[e]);
        float p[NEXP];
        for (int e = 0; e < NEXP; ++e) p[e] = __expf(acc[e] - mx);
        int i0 = 0;
        for (int e = 1; e < NEXP; ++e) if (p[e] > p[i0]) i0 = e;
        int i1 = (i0 == 0) ? 1 : 0;
        for (int e = 0; e < NEXP; ++e) if (e != i0 && p[e] > p[i1]) i1 = e;
        // softmax denominator cancels in the renormalized top-2 weights
        const float s2 = p[i0] + p[i1];
        const float w0 = p[i0] / s2;
        const float w1 = p[i1] / s2;
        int pos = atomicAdd(&cnt[i0], 1);
        toks[i0 * T + pos] = t; wts[i0 * T + pos] = w0;
        pos = atomicAdd(&cnt[i1], 1);
        toks[i1 * T + pos] = t; wts[i1 * T + pos] = w1;
    }
}

// ------------- weight convert+transpose: f32 [E][R][C] -> f16 [E][C][R] -------------
__global__ __launch_bounds__(256) void convert_transpose_kernel(
    const float* __restrict__ W, _Float16* __restrict__ Wt, int R, int C)
{
    __shared__ float tile[32][33];
    const int e  = blockIdx.z;
    const int c0 = blockIdx.x * 32;
    const int r0 = blockIdx.y * 32;
    const float* We  = W  + (size_t)e * R * C;
    _Float16*    Wte = Wt + (size_t)e * R * C;
    const int tx = threadIdx.x, ty = threadIdx.y;
    for (int i = ty; i < 32; i += 8)
        tile[i][tx] = We[(size_t)(r0 + i) * C + (c0 + tx)];
    __syncthreads();
    for (int i = ty; i < 32; i += 8)
        Wte[(size_t)(c0 + i) * R + (r0 + tx)] = (_Float16)tile[tx][i];
}

// ---------------- fused per-expert FFN: gelu(x@W1+b1)@W2+b2, scaled scatter ----------------
// 8 waves = 2 row-groups (16 tokens each) x 4 col-groups.
__global__ __launch_bounds__(256) void moe_ffn_kernel(
    const float*    __restrict__ x,     // [T][D] f32
    const _Float16* __restrict__ W1t,   // [E][H][D] f16 (transposed)
    const float*    __restrict__ b1,    // [E][H]
    const _Float16* __restrict__ W2t,   // [E][D][H] f16 (transposed)
    const float*    __restrict__ b2,    // [E][D]
    const int*      __restrict__ cnt,   // [E]
    const int*      __restrict__ toks,  // [E][T]
    const float*    __restrict__ wts,   // [E][T]
    float*          __restrict__ out,   // [T][D]
    int T)
{
    extern __shared__ _Float16 smem[];
    _Float16* xs = smem;                          // [MT][D_DIM]
    _Float16* hs = smem + MT * D_DIM;             // [MT][HC]
    int*   s_tok = (int*)(smem + MT * D_DIM + MT * HC);
    float* s_w   = (float*)(s_tok + MT);

    const int e    = blockIdx.y;
    const int ntok = cnt[e];
    const int row0 = blockIdx.x * MT;
    if (row0 >= ntok) return;           // uniform across block

    const int tid  = threadIdx.x;
    const int wave = tid >> 5;
    const int lane = tid & 31;
    const int l15  = lane & 15;
    const int kh   = lane >> 4;
    const int rowg = wave >> 2;         // 0..1 : token rows [rowg*16, +16)
    const int colg = wave & 3;          // 0..3 : column group

    if (tid < MT) {
        const int r = row0 + tid;
        s_tok[tid] = (r < ntok) ? toks[e * T + r] : -1;
        s_w[tid]   = (r < ntok) ? wts[e * T + r] : 0.0f;
    }
    __syncthreads();

    // stage x rows (f32 -> f16) into LDS, zero padding rows
    for (int idx = tid; idx < MT * D_DIM; idx += 256) {
        const int r = idx >> 10;
        const int c = idx & (D_DIM - 1);
        const int t = s_tok[r];
        xs[idx] = (_Float16)((t >= 0) ? x[(size_t)t * D_DIM + c] : 0.0f);
    }
    __syncthreads();

    const _Float16* W1e = W1t + (size_t)e * H_DIM * D_DIM;
    const _Float16* W2e = W2t + (size_t)e * D_DIM * H_DIM;
    const float* b1e = b1 + e * H_DIM;
    const float* b2e = b2 + e * D_DIM;

    // phase-2 accumulators: wave owns rows [rowg*16,+16) x cols [colg*256,+256)
    v8f acc[16];
#pragma unroll
    for (int j = 0; j < 16; ++j)
#pragma unroll
        for (int q = 0; q < 8; ++q) acc[j][q] = 0.0f;

    // phase-2 B base: column d = colg*256 + j*16 + l15; j folds into immediate offsets
    const _Float16* B2base = W2e + (size_t)(colg * 256 + l15) * H_DIM + kh * 16;
    const int am = rowg * 16 + l15;     // A-fragment row for this lane

    for (int h0 = 0; h0 < H_DIM; h0 += HC) {
        // ---- phase 1: hs[MT][HC] = gelu(xs @ W1e[:, h0:h0+HC] + b1)
        v8f c4[4];
#pragma unroll
        for (int j = 0; j < 4; ++j)
#pragma unroll
            for (int q = 0; q < 8; ++q) c4[j][q] = 0.0f;

        // B column h = h0 + colg*64 + j*16 + l15
        const _Float16* B1base = W1e + (size_t)(h0 + colg * 64 + l15) * D_DIM + kh * 16;

        for (int k0 = 0; k0 < D_DIM; k0 += 32) {
            v16h a = load_a_frag(xs, D_DIM, am, kh, k0);
#pragma unroll
            for (int j = 0; j < 4; ++j) {
                const _Float16* Bp = B1base + (size_t)j * (16 * D_DIM) + k0;
                __builtin_prefetch(Bp + 256, 0, 1);            // global_prefetch_b8
                v16h b = *(const v16h*)Bp;                     // 32B contiguous
                c4[j] = __builtin_amdgcn_wmma_f32_16x16x32_f16(
                            false, a, false, b, (short)0, c4[j], false, false);
            }
        }
#pragma unroll
        for (int j = 0; j < 4; ++j) {
            const int nl = colg * 64 + j * 16;
            const float bias = b1e[h0 + nl + l15];
#pragma unroll
            for (int r = 0; r < 8; ++r)
                hs[(rowg * 16 + kh * 8 + r) * HC + nl + l15] =
                    (_Float16)gelu_f(c4[j][r] + bias);
        }
        __syncthreads();

        // ---- phase 2: acc += hs @ W2e[h0:h0+HC, :]; 16 independent accumulators
        for (int k0 = 0; k0 < HC; k0 += 32) {
            v16h a = load_a_frag(hs, HC, am, kh, k0);
            const int kg = h0 + k0;
#pragma unroll
            for (int j = 0; j < 16; ++j) {
                const _Float16* Bp = B2base + (size_t)j * (16 * H_DIM) + kg;
                __builtin_prefetch(Bp + 256, 0, 1);
                v16h b = *(const v16h*)Bp;
                acc[j] = __builtin_amdgcn_wmma_f32_16x16x32_f16(
                             false, a, false, b, (short)0, acc[j], false, false);
            }
        }
        __syncthreads();
    }

    // epilogue: bias + routing weight, scatter-add (exactly 2 adds/element -> deterministic)
#pragma unroll
    for (int j = 0; j < 16; ++j) {
        const int nd = colg * 256 + j * 16 + l15;
        const float bias2 = b2e[nd];
#pragma unroll
        for (int r = 0; r < 8; ++r) {
            const int m = rowg * 16 + kh * 8 + r;
            const int t = s_tok[m];
            if (t >= 0) {
                const float v = (acc[j][r] + bias2) * s_w[m];
                __hip_atomic_fetch_add(&out[(size_t)t * D_DIM + nd], v,
                                       __ATOMIC_RELAXED, __HIP_MEMORY_SCOPE_AGENT);
            }
        }
    }
}

extern "C" void kernel_launch(void* const* d_in, const int* in_sizes, int n_in,
                              void* d_out, int out_size, void* d_ws, size_t ws_size,
                              hipStream_t stream) {
    const float* x  = (const float*)d_in[0];
    const float* Wg = (const float*)d_in[1];
    const float* W1 = (const float*)d_in[2];
    const float* b1 = (const float*)d_in[3];
    const float* W2 = (const float*)d_in[4];
    const float* b2 = (const float*)d_in[5];
    float* out = (float*)d_out;
    const int T = in_sizes[0] / D_DIM;      // B*S

    size_t off = 0;
    auto carve = [&](size_t bytes) -> void* {
        void* p = (char*)d_ws + off;
        off += (bytes + 255) & ~(size_t)255;
        return p;
    };
    _Float16* W1t = (_Float16*)carve((size_t)NEXP * H_DIM * D_DIM * sizeof(_Float16)); // 64 MB
    _Float16* W2t = (_Float16*)carve((size_t)NEXP * H_DIM * D_DIM * sizeof(_Float16)); // 64 MB
    int*   cnt  = (int*)carve(NEXP * sizeof(int));
    int*   toks = (int*)carve((size_t)NEXP * T * sizeof(int));
    float* wtb  = (float*)carve((size_t)NEXP * T * sizeof(float));
    (void)ws_size; (void)n_in; (void)out_size;

    hipMemsetAsync(d_out, 0, (size_t)T * D_DIM * sizeof(float), stream);
    hipMemsetAsync(cnt, 0, NEXP * sizeof(int), stream);

    {   // W1 [E][D][H] -> W1t [E][H][D]
        dim3 g(H_DIM / 32, D_DIM / 32, NEXP), b(32, 8);
        convert_transpose_kernel<<<g, b, 0, stream>>>(W1, W1t, D_DIM, H_DIM);
    }
    {   // W2 [E][H][D] -> W2t [E][D][H]
        dim3 g(D_DIM / 32, H_DIM / 32, NEXP), b(32, 8);
        convert_transpose_kernel<<<g, b, 0, stream>>>(W2, W2t, H_DIM, D_DIM);
    }

    moe_gate_kernel<<<dim3((T + NW - 1) / NW), dim3(256), 0, stream>>>(
        x, Wg, cnt, toks, wtb, T);

    moe_ffn_kernel<<<dim3((T + MT - 1) / MT, NEXP), dim3(256), SMEM_BYTES, stream>>>(
        x, W1t, b1, W2t, b2, cnt, toks, wtb, out, T);
}
